// HexagdlyNet_18038862643938
// MI455X (gfx1250) — compile-verified, matrix operand order checked by reading, not run on hardware
//
#include <hip/hip_runtime.h>
#include <hip/hip_bf16.h>

// ---------------------------------------------------------------------------
// HexagdlyNet for MI455X (gfx1250, wave32).
// Convs: tiny channel counts -> fused VALU kernels, f16 activations.
// FC1/2/3: v_wmma_f32_16x16x32_f16; LDS-staged A panel shared by 8 waves,
// 16x64 output tile per wave (A fragment reused across 4 WMMAs), fully
// branchless inner loop (no EXEC manipulation around WMMA).
// ---------------------------------------------------------------------------

typedef _Float16 v16h __attribute__((ext_vector_type(16)));
typedef _Float16 v8h  __attribute__((ext_vector_type(8)));
typedef float    v8f  __attribute__((ext_vector_type(8)));

#define BN_EPS 1e-5f

// ---------------------------------------------------------------------------
// f32 -> f16 conversion (FC weights)
// ---------------------------------------------------------------------------
__global__ void cvt_f32_to_f16(const float* __restrict__ in,
                               _Float16* __restrict__ out, int n) {
  int i = blockIdx.x * blockDim.x + threadIdx.x;
  if (i < n) out[i] = (_Float16)in[i];
}

// ---------------------------------------------------------------------------
// Layer 0: hexconv n=2 (Cin=1, Cout=8) + BN + ReLU.  Input f32 image.
// Output (B,8,40,32) f16.
// ---------------------------------------------------------------------------
__global__ __launch_bounds__(256)
void hexconv0_bn_relu(const float* __restrict__ img,
                      const float* __restrict__ w0,  // (8,1,5,1)
                      const float* __restrict__ w1,  // (8,1,4,2)
                      const float* __restrict__ w2,  // (8,1,3,2)
                      const float* __restrict__ bias,
                      const float* __restrict__ g, const float* __restrict__ bb,
                      const float* __restrict__ bm, const float* __restrict__ bv,
                      _Float16* __restrict__ out, int B) {
  const int H = 40, W = 32;
  int idx = blockIdx.x * blockDim.x + threadIdx.x;
  int c = idx % W;
  int r = (idx / W) % H;
  int b = idx / (W * H);
  if (b >= B) return;

  const float* im = img + (size_t)b * H * W;
  auto at = [&](int rr, int cc) -> float {
    return (rr >= 0 && rr < H && cc >= 0 && cc < W) ? im[rr * W + cc] : 0.f;
  };

  float acc[8];
#pragma unroll
  for (int co = 0; co < 8; ++co) acc[co] = bias[co];

  // i=0: center column, rows r-2..r+2
#pragma unroll
  for (int kr = 0; kr < 5; ++kr) {
    float x = at(r - 2 + kr, c);
#pragma unroll
    for (int co = 0; co < 8; ++co) acc[co] += w0[co * 5 + kr] * x;
  }
  // i=1: cols c+-1, 4 rows; even col rows r-2..r+1, odd col rows r-1..r+2
  int rb1 = r - 2 + (c & 1);
#pragma unroll
  for (int kr = 0; kr < 4; ++kr) {
    float xm = at(rb1 + kr, c - 1);
    float xp = at(rb1 + kr, c + 1);
#pragma unroll
    for (int co = 0; co < 8; ++co) {
      acc[co] += w1[co * 8 + kr * 2 + 0] * xm + w1[co * 8 + kr * 2 + 1] * xp;
    }
  }
  // i=2: cols c+-2, rows r-1..r+1 (both parities)
#pragma unroll
  for (int kr = 0; kr < 3; ++kr) {
    float xm = at(r - 1 + kr, c - 2);
    float xp = at(r - 1 + kr, c + 2);
#pragma unroll
    for (int co = 0; co < 8; ++co) {
      acc[co] += w2[co * 6 + kr * 2 + 0] * xm + w2[co * 6 + kr * 2 + 1] * xp;
    }
  }
#pragma unroll
  for (int co = 0; co < 8; ++co) {
    float s = g[co] * rsqrtf(bv[co] + BN_EPS);
    float y = (acc[co] - bm[co]) * s + bb[co];
    y = fmaxf(y, 0.f);
    out[((size_t)(b * 8 + co) * H + r) * W + c] = (_Float16)y;
  }
}

// ---------------------------------------------------------------------------
// hexconv n=1 (CIN->COUT) + BN + ReLU.  f16 in/out.
// ---------------------------------------------------------------------------
template <int CIN, int COUT>
__global__ __launch_bounds__(256)
void hexconv1_bn_relu(const _Float16* __restrict__ in,
                      const float* __restrict__ w0,  // (COUT,CIN,3,1)
                      const float* __restrict__ w1,  // (COUT,CIN,2,2)
                      const float* __restrict__ bias,
                      const float* __restrict__ g, const float* __restrict__ bb,
                      const float* __restrict__ bm, const float* __restrict__ bv,
                      _Float16* __restrict__ out, int B, int H, int W) {
  int idx = blockIdx.x * blockDim.x + threadIdx.x;
  int c = idx % W;
  int r = (idx / W) % H;
  int b = idx / (W * H);
  if (b >= B) return;

  float acc[COUT];
#pragma unroll
  for (int co = 0; co < COUT; ++co) acc[co] = bias[co];

  int rb = r - 1 + (c & 1);  // i=1 row window: even col r-1..r, odd col r..r+1

  for (int ci = 0; ci < CIN; ++ci) {
    const _Float16* pc = in + ((size_t)(b * CIN + ci) * H) * W;
    auto at = [&](int rr, int cc) -> float {
      return (rr >= 0 && rr < H && cc >= 0 && cc < W) ? (float)pc[rr * W + cc]
                                                      : 0.f;
    };
    float t0 = at(r - 1, c), t1 = at(r, c), t2 = at(r + 1, c);
    float lm0 = at(rb, c - 1), lm1 = at(rb + 1, c - 1);
    float lp0 = at(rb, c + 1), lp1 = at(rb + 1, c + 1);
#pragma unroll
    for (int co = 0; co < COUT; ++co) {
      const float* W0 = w0 + (co * CIN + ci) * 3;
      const float* W1 = w1 + (co * CIN + ci) * 4;
      acc[co] += W0[0] * t0 + W0[1] * t1 + W0[2] * t2 + W1[0] * lm0 +
                 W1[1] * lp0 + W1[2] * lm1 + W1[3] * lp1;
    }
  }
#pragma unroll
  for (int co = 0; co < COUT; ++co) {
    float s = g[co] * rsqrtf(bv[co] + BN_EPS);
    float y = (acc[co] - bm[co]) * s + bb[co];
    y = fmaxf(y, 0.f);
    out[((size_t)(b * COUT + co) * H + r) * W + c] = (_Float16)y;
  }
}

// ---------------------------------------------------------------------------
// hexpool stride 2.  Subsampled positions always land on even input columns,
// so the "even" 7-cell neighbor set applies everywhere.  Inputs >= 0, so the
// zero-pad is implemented by the 0 init.  f16 in/out.
// ---------------------------------------------------------------------------
__global__ __launch_bounds__(256)
void hexpool_s2(const _Float16* __restrict__ in, _Float16* __restrict__ out,
                int B, int C, int H, int W) {
  int OH = H / 2, OW = W / 2;
  int idx = blockIdx.x * blockDim.x + threadIdx.x;
  int cw = idx % OW;
  int ro = (idx / OW) % OH;
  int ch = (idx / (OW * OH)) % C;
  int b = idx / (OW * OH * C);
  if (b >= B) return;

  int r = 2 * ro + (cw & 1);
  int c = 2 * cw;
  const _Float16* pc = in + ((size_t)(b * C + ch) * H) * W;
  auto at = [&](int rr, int cc) -> float {
    return (rr >= 0 && rr < H && cc >= 0 && cc < W) ? (float)pc[rr * W + cc]
                                                    : 0.f;
  };
  float m = at(r, c);
  m = fmaxf(m, at(r - 1, c));
  m = fmaxf(m, at(r + 1, c));
  m = fmaxf(m, at(r - 1, c - 1));
  m = fmaxf(m, at(r, c - 1));
  m = fmaxf(m, at(r - 1, c + 1));
  m = fmaxf(m, at(r, c + 1));
  out[((size_t)(b * C + ch) * OH + ro) * OW + cw] = (_Float16)m;
}

// ---------------------------------------------------------------------------
// hexpool stride 1 + flatten into the concatenated FC input (f16).
// C=32, H=10, W=8 -> h0[b][branch*2560 + ch*80 + r*8 + c]
// ---------------------------------------------------------------------------
__global__ __launch_bounds__(256)
void hexpool_s1_flat(const _Float16* __restrict__ in, _Float16* __restrict__ h0,
                     int B, int branch) {
  const int C = 32, H = 10, W = 8;
  int idx = blockIdx.x * blockDim.x + threadIdx.x;
  int c = idx % W;
  int r = (idx / W) % H;
  int ch = (idx / (W * H)) % C;
  int b = idx / (W * H * C);
  if (b >= B) return;

  const _Float16* pc = in + ((size_t)(b * C + ch) * H) * W;
  auto at = [&](int rr, int cc) -> float {
    return (rr >= 0 && rr < H && cc >= 0 && cc < W) ? (float)pc[rr * W + cc]
                                                    : 0.f;
  };
  float m = at(r, c);
  m = fmaxf(m, at(r - 1, c));
  m = fmaxf(m, at(r + 1, c));
  if ((c & 1) == 0) {  // even column
    m = fmaxf(m, at(r - 1, c - 1));
    m = fmaxf(m, at(r, c - 1));
    m = fmaxf(m, at(r - 1, c + 1));
    m = fmaxf(m, at(r, c + 1));
  } else {  // odd column
    m = fmaxf(m, at(r, c - 1));
    m = fmaxf(m, at(r + 1, c - 1));
    m = fmaxf(m, at(r, c + 1));
    m = fmaxf(m, at(r + 1, c + 1));
  }
  h0[(size_t)b * 5120 + branch * 2560 + ch * 80 + r * 8 + c] = (_Float16)m;
}

// ---------------------------------------------------------------------------
// WMMA GEMM: Y[M,N] = relu(X[M,K] * Wt[N,K]^T + bias[N]), f16 in/out,
// f32 accumulate via v_wmma_f32_16x16x32_f16.
//
// Block = 256 threads = 8 waves, one 16-row M tile per block.  The 16xK A
// panel is staged through LDS in K-chunks of 256 (8.25 KB with bank padding)
// and shared by all 8 waves; each wave owns a 16x64 output tile (4 WMMA
// accumulators) so the A fragment fetched from LDS is reused 4x from
// registers while B streams from global (L2-resident; 192 MB L2 >> weights).
//
// All N and per-wave N-bases are multiples of 64, so a wave is either fully
// in-range or fully out-of-range; out-of-range waves run the (EXEC-all-ones,
// branchless) WMMAs on clamped B rows and only skip the epilogue stores.
// This keeps the inner loop free of saveexec/divergence so accumulators stay
// pinned in their WMMA register quads.
//
// Fragment layout (ISA 7.12.2, 16-bit A 16x32): lane L<16 holds row L with
// K 0..7 / 16..23 in v0..7, lane L+16 holds K 8..15 / 24..31; B mirrors with
// N in place of M, matching row-major (N,K) weight storage.  LDS row stride
// is 264 halfs (528 B) so consecutive rows start 4 banks apart -> the 16-lane
// ds_load_b128 fragment fetch is bank-conflict-free.
// ---------------------------------------------------------------------------
#define LDS_K 256
#define LDS_STRIDE 264

__global__ __launch_bounds__(256)
void fc_wmma(const _Float16* __restrict__ X, const _Float16* __restrict__ Wt,
             const float* __restrict__ bias, _Float16* __restrict__ Y, int M,
             int N, int K, int relu) {
  __shared__ _Float16 sA[16 * LDS_STRIDE];

  const int wave = threadIdx.x >> 5;
  const int lane = threadIdx.x & 31;
  const int half = lane >> 4;
  const int lr = lane & 15;
  const int m0 = blockIdx.y * 16;
  const int nw = blockIdx.x * 512 + wave * 64;  // this wave's N base
  const int nwc = min(nw, N - 64);              // clamped (wave-uniform)

  const _Float16* brow0 = Wt + (size_t)(nwc + 0 * 16 + lr) * K + half * 8;
  const _Float16* brow1 = Wt + (size_t)(nwc + 1 * 16 + lr) * K + half * 8;
  const _Float16* brow2 = Wt + (size_t)(nwc + 2 * 16 + lr) * K + half * 8;
  const _Float16* brow3 = Wt + (size_t)(nwc + 3 * 16 + lr) * K + half * 8;

  v8f acc0 = {0.f, 0.f, 0.f, 0.f, 0.f, 0.f, 0.f, 0.f};
  v8f acc1 = acc0, acc2 = acc0, acc3 = acc0;

  // Cooperative staging map: thread -> (row, 16-half column slice)
  const int srow = threadIdx.x >> 4;         // 0..15
  const int scol = (threadIdx.x & 15) * 16;  // 0,16,...,240
  const _Float16* sgrow = X + (size_t)(m0 + srow) * K + scol;

  for (int k0 = 0; k0 < K; k0 += LDS_K) {
    __syncthreads();
    {  // stage 16 x 256 A chunk: two b128 global loads -> two b128 LDS stores
      v8h s0 = *(const v8h*)(sgrow + k0);
      v8h s1 = *(const v8h*)(sgrow + k0 + 8);
      *(v8h*)(&sA[srow * LDS_STRIDE + scol]) = s0;
      *(v8h*)(&sA[srow * LDS_STRIDE + scol + 8]) = s1;
    }
    __syncthreads();

#pragma unroll 2
    for (int kk = 0; kk < LDS_K; kk += 32) {
      v8h a_lo = *(const v8h*)(&sA[lr * LDS_STRIDE + kk + half * 8]);
      v8h a_hi = *(const v8h*)(&sA[lr * LDS_STRIDE + kk + half * 8 + 16]);
      v16h a = __builtin_shufflevector(a_lo, a_hi, 0, 1, 2, 3, 4, 5, 6, 7, 8,
                                       9, 10, 11, 12, 13, 14, 15);

      v8h b_lo, b_hi;
      v16h b;

      b_lo = *(const v8h*)(brow0 + k0 + kk);
      b_hi = *(const v8h*)(brow0 + k0 + kk + 16);
      b = __builtin_shufflevector(b_lo, b_hi, 0, 1, 2, 3, 4, 5, 6, 7, 8, 9, 10,
                                  11, 12, 13, 14, 15);
      acc0 = __builtin_amdgcn_wmma_f32_16x16x32_f16(false, a, false, b,
                                                    (short)0, acc0, false, false);

      b_lo = *(const v8h*)(brow1 + k0 + kk);
      b_hi = *(const v8h*)(brow1 + k0 + kk + 16);
      b = __builtin_shufflevector(b_lo, b_hi, 0, 1, 2, 3, 4, 5, 6, 7, 8, 9, 10,
                                  11, 12, 13, 14, 15);
      acc1 = __builtin_amdgcn_wmma_f32_16x16x32_f16(false, a, false, b,
                                                    (short)0, acc1, false, false);

      b_lo = *(const v8h*)(brow2 + k0 + kk);
      b_hi = *(const v8h*)(brow2 + k0 + kk + 16);
      b = __builtin_shufflevector(b_lo, b_hi, 0, 1, 2, 3, 4, 5, 6, 7, 8, 9, 10,
                                  11, 12, 13, 14, 15);
      acc2 = __builtin_amdgcn_wmma_f32_16x16x32_f16(false, a, false, b,
                                                    (short)0, acc2, false, false);

      b_lo = *(const v8h*)(brow3 + k0 + kk);
      b_hi = *(const v8h*)(brow3 + k0 + kk + 16);
      b = __builtin_shufflevector(b_lo, b_hi, 0, 1, 2, 3, 4, 5, 6, 7, 8, 9, 10,
                                  11, 12, 13, 14, 15);
      acc3 = __builtin_amdgcn_wmma_f32_16x16x32_f16(false, a, false, b,
                                                    (short)0, acc3, false, false);
    }
  }

  // Epilogue: D layout (VGPR j -> row m0+8*half+j, lane lr -> column).
  // Stores are per-lane; EXEC divergence is fine here (no WMMA below).
  if (nw < N) {
    v8f accs[4] = {acc0, acc1, acc2, acc3};
#pragma unroll
    for (int t = 0; t < 4; ++t) {
      int nn = nw + t * 16 + lr;
      float bv = bias[nn];
      _Float16* ycol = Y + (size_t)(m0 + half * 8) * N + nn;
#pragma unroll
      for (int j = 0; j < 8; ++j) {
        float v = accs[t][j] + bv;
        if (relu) v = fmaxf(v, 0.f);
        ycol[(size_t)j * N] = (_Float16)v;
      }
    }
  }
}

// ---------------------------------------------------------------------------
// Final FC: (1024,64) -> (1024,2), f32 output, no relu.
// ---------------------------------------------------------------------------
__global__ __launch_bounds__(256)
void fc_last(const _Float16* __restrict__ X, const float* __restrict__ W4,
             const float* __restrict__ b4, float* __restrict__ out, int B) {
  int idx = blockIdx.x * blockDim.x + threadIdx.x;
  if (idx >= B * 2) return;
  int b = idx >> 1;
  int o = idx & 1;
  const _Float16* x = X + (size_t)b * 64;
  const float* w = W4 + o * 64;
  float s = b4[o];
#pragma unroll
  for (int k = 0; k < 64; ++k) s += (float)x[k] * w[k];
  out[idx] = s;
}

// ---------------------------------------------------------------------------
// Launch
// ---------------------------------------------------------------------------
extern "C" void kernel_launch(void* const* d_in, const int* in_sizes, int n_in,
                              void* d_out, int out_size, void* d_ws,
                              size_t ws_size, hipStream_t stream) {
  const int B = 1024;
  char* base = (char*)d_ws;
  size_t off = 0;
  auto alloc = [&](size_t bytes) -> char* {
    off = (off + 255) & ~(size_t)255;
    char* p = base + off;
    off += bytes;
    return p;
  };

  _Float16* h0  = (_Float16*)alloc((size_t)B * 5120 * 2);
  _Float16* w1h = (_Float16*)alloc((size_t)1024 * 5120 * 2);
  _Float16* w2h = (_Float16*)alloc((size_t)256 * 1024 * 2);
  _Float16* w3h = (_Float16*)alloc((size_t)64 * 256 * 2);
  _Float16* y1  = (_Float16*)alloc((size_t)B * 1024 * 2);
  _Float16* y2  = (_Float16*)alloc((size_t)B * 256 * 2);
  _Float16* y3  = (_Float16*)alloc((size_t)B * 64 * 2);
  _Float16* act0 = (_Float16*)alloc((size_t)B * 8 * 40 * 32 * 2);
  _Float16* p0   = (_Float16*)alloc((size_t)B * 8 * 20 * 16 * 2);
  _Float16* act1 = (_Float16*)alloc((size_t)B * 16 * 20 * 16 * 2);
  _Float16* p1   = (_Float16*)alloc((size_t)B * 16 * 10 * 8 * 2);
  _Float16* act2 = (_Float16*)alloc((size_t)B * 32 * 10 * 8 * 2);

  auto F = [&](int i) { return (const float*)d_in[i]; };

  // FC weight conversions (d_in: 3=fc0.b 4=fc0.w 5=fc1.b 6=fc1.w 7=fc2.b 8=fc2.w
  //                               9=fc3.b 10=fc3.w)
  cvt_f32_to_f16<<<(1024 * 5120 + 255) / 256, 256, 0, stream>>>(F(4), w1h,
                                                                1024 * 5120);
  cvt_f32_to_f16<<<(256 * 1024 + 255) / 256, 256, 0, stream>>>(F(6), w2h,
                                                               256 * 1024);
  cvt_f32_to_f16<<<(64 * 256 + 255) / 256, 256, 0, stream>>>(F(8), w3h,
                                                             64 * 256);

  for (int br = 0; br < 2; ++br) {
    const float* img = F(br);          // m1_image / m2_image
    int bnb = 11 + br * 22;            // bn leaves: bb,g,m,v per layer
    int cb = 23 + br * 22;             // conv leaves: b,w0,w1[,w2] per layer

    hexconv0_bn_relu<<<(B * 40 * 32 + 255) / 256, 256, 0, stream>>>(
        img, F(cb + 1), F(cb + 2), F(cb + 3), F(cb + 0), F(bnb + 1),
        F(bnb + 0), F(bnb + 2), F(bnb + 3), act0, B);

    hexpool_s2<<<(B * 8 * 20 * 16 + 255) / 256, 256, 0, stream>>>(act0, p0, B,
                                                                  8, 40, 32);

    hexconv1_bn_relu<8, 16><<<(B * 20 * 16 + 255) / 256, 256, 0, stream>>>(
        p0, F(cb + 5), F(cb + 6), F(cb + 4), F(bnb + 5), F(bnb + 4),
        F(bnb + 6), F(bnb + 7), act1, B, 20, 16);

    hexpool_s2<<<(B * 16 * 10 * 8 + 255) / 256, 256, 0, stream>>>(act1, p1, B,
                                                                  16, 20, 16);

    hexconv1_bn_relu<16, 32><<<(B * 10 * 8 + 255) / 256, 256, 0, stream>>>(
        p1, F(cb + 8), F(cb + 9), F(cb + 7), F(bnb + 9), F(bnb + 8),
        F(bnb + 10), F(bnb + 11), act2, B, 10, 8);

    hexpool_s1_flat<<<(B * 32 * 10 * 8 + 255) / 256, 256, 0, stream>>>(
        act2, h0, B, br);
  }

  // FC stack: block = 16 M-rows x 512 N-cols (8 waves x 4 WMMA tiles)
  fc_wmma<<<dim3((1024 + 511) / 512, 64), 256, 0, stream>>>(h0, w1h, F(3), y1,
                                                            B, 1024, 5120, 1);
  fc_wmma<<<dim3((256 + 511) / 512, 64), 256, 0, stream>>>(y1, w2h, F(5), y2,
                                                           B, 256, 1024, 1);
  fc_wmma<<<dim3((64 + 511) / 512, 64), 256, 0, stream>>>(y2, w3h, F(7), y3,
                                                          B, 64, 256, 1);

  fc_last<<<(B * 2 + 255) / 256, 256, 0, stream>>>(y3, F(10), F(9),
                                                   (float*)d_out, B);
}
